// SOMLayer_32899449487560
// MI455X (gfx1250) — matrix operand order repaired and checked by reading, not compiled, not run on hardware
//
#include <hip/hip_runtime.h>

typedef _Float16 v8h  __attribute__((ext_vector_type(8)));
typedef _Float16 v16h __attribute__((ext_vector_type(16)));
typedef float    v8f  __attribute__((ext_vector_type(8)));

namespace {
constexpr int   kD = 256;          // latent dim
constexpr int   kK = 1024;         // num nodes
constexpr int   kN = 32768;        // 64*512 rows
constexpr float kInvAlpha = 0.1f;  // 1/alpha
constexpr float kExp = -5.5f;      // -(alpha+1)/2
}

union V16U { v16h v; v8h h[2]; };

// x >= 1 always here, so pow has no edge cases: x^e = exp2(e * log2(x)).
// Lowers to v_log_f32 + v_mul_f32 + v_exp_f32.
__device__ __forceinline__ float fast_pow_ge1(float x, float e) {
  return __builtin_amdgcn_exp2f(e * __builtin_amdgcn_logf(x));
}

// Split f32 -> (f16 hi, f16 lo) so hi+lo reproduces f32 to ~2^-21.
__global__ void cvt_split_kernel(const float* __restrict__ x,
                                 _Float16* __restrict__ hi,
                                 _Float16* __restrict__ lo, int n) {
  int i = blockIdx.x * blockDim.x + threadIdx.x;
  if (i < n) {
    float v = x[i];
    _Float16 h = (_Float16)v;
    hi[i] = h;
    lo[i] = (_Float16)(v - (float)h);
  }
}

// Exact f32 ||c_k||^2, computed once for all blocks.
__global__ void csq_kernel(const float* __restrict__ centers,
                           float* __restrict__ csq) {
  int k = blockIdx.x * blockDim.x + threadIdx.x;
  if (k < kK) {
    const float* row = centers + (size_t)k * kD;
    float s = 0.f;
    #pragma unroll 4
    for (int i = 0; i < kD; ++i) s = fmaf(row[i], row[i], s);
    csq[k] = s;
  }
}

// One block = 16 rows x 1024 cols. 8 waves, each wave owns 8 16x16 tiles.
__global__ __launch_bounds__(256)
void som_soft_assign_kernel(const float* __restrict__ z,
                            const _Float16* __restrict__ zh,
                            const _Float16* __restrict__ zl,
                            const _Float16* __restrict__ ch,
                            const _Float16* __restrict__ cl,
                            const float* __restrict__ csq,
                            float* __restrict__ out) {
  __shared__ float s_part[16][16];
  __shared__ float s_zsq[16];
  __shared__ float s_rowpart[16][8];
  __shared__ float s_inv[16];

  const int tid  = threadIdx.x;
  const int wave = tid >> 5;
  const int lane = tid & 31;
  const int lh   = lane >> 4;   // half-wave select
  const int lm   = lane & 15;   // M (A/C) or N (B/C) index
  const int m0   = blockIdx.x * 16;

  // ---- exact f32 ||z_m||^2 for this 16-row block ----
  {
    const int row = tid >> 4;
    const int j   = tid & 15;
    const float* zr = z + (size_t)(m0 + row) * kD;
    float s = 0.f;
    for (int i = j; i < kD; i += 16) { float v = zr[i]; s = fmaf(v, v, s); }
    s_part[row][j] = s;
  }
  __syncthreads();
  if (tid < 16) {
    float s = 0.f;
    #pragma unroll
    for (int j = 0; j < 16; ++j) s += s_part[tid][j];
    s_zsq[tid] = s;
  }
  __syncthreads();

  const int n_base = wave * 128;

  float csqv[8];
  #pragma unroll
  for (int t = 0; t < 8; ++t) csqv[t] = csq[n_base + t * 16 + lm];

  float zsq_loc[8];
  #pragma unroll
  for (int r = 0; r < 8; ++r) zsq_loc[r] = s_zsq[r + lh * 8];

  v8f acc[8];
  #pragma unroll
  for (int t = 0; t < 8; ++t)
    acc[t] = (v8f){0.f, 0.f, 0.f, 0.f, 0.f, 0.f, 0.f, 0.f};

  // A rows (z) and B columns (centers), f16 hi/lo, contiguous k = 16h + s for both A and B.
  const _Float16* zhr = zh + (size_t)(m0 + lm) * kD;
  const _Float16* zlr = zl + (size_t)(m0 + lm) * kD;
  const _Float16* chb = ch + (size_t)(n_base + lm) * kD;
  const _Float16* clb = cl + (size_t)(n_base + lm) * kD;

  for (int c = 0; c < 8; ++c) {            // 8 chunks of K=32 over D=256
    const int aoff = c * 32 + lh * 16;     // A 16-bit 16x32 layout, contiguous k = 16h + s
    V16U ah, al;
    ah.h[0] = *(const v8h*)(zhr + aoff);
    ah.h[1] = *(const v8h*)(zhr + aoff + 8);
    al.h[0] = *(const v8h*)(zlr + aoff);
    al.h[1] = *(const v8h*)(zlr + aoff + 8);
    const int boff = c * 32 + lh * 16;     // B 16-bit 32x16 layout
    #pragma unroll
    for (int t = 0; t < 8; ++t) {
      V16U bh, bl;
      bh.h[0] = *(const v8h*)(chb + t * (16 * kD) + boff);
      bh.h[1] = *(const v8h*)(chb + t * (16 * kD) + boff + 8);
      bl.h[0] = *(const v8h*)(clb + t * (16 * kD) + boff);
      bl.h[1] = *(const v8h*)(clb + t * (16 * kD) + boff + 8);
      // cross = zh*ch + zl*ch + zh*cl  (split-f16 emulated f32)
      acc[t] = __builtin_amdgcn_wmma_f32_16x16x32_f16(
          false, ah.v, false, bh.v, (short)0, acc[t], false, false);
      acc[t] = __builtin_amdgcn_wmma_f32_16x16x32_f16(
          false, al.v, false, bh.v, (short)0, acc[t], false, false);
      acc[t] = __builtin_amdgcn_wmma_f32_16x16x32_f16(
          false, ah.v, false, bl.v, (short)0, acc[t], false, false);
    }
  }

  // ---- fused epilogue: Student-t kernel, in registers ----
  float partial[8];
  #pragma unroll
  for (int r = 0; r < 8; ++r) partial[r] = 0.f;

  #pragma unroll
  for (int t = 0; t < 8; ++t) {
    #pragma unroll
    for (int r = 0; r < 8; ++r) {
      float cross = acc[t][r];
      float dd = zsq_loc[r] + csqv[t] - 2.f * cross;
      dd = fmaxf(dd, 0.f);
      float sv = fast_pow_ge1(fmaf(dd, kInvAlpha, 1.f), kExp);
      acc[t][r] = sv;
      partial[r] += sv;
    }
  }

  // reduce over the 16 lanes of each half-wave (rows split by lh)
  #pragma unroll
  for (int r = 0; r < 8; ++r) {
    #pragma unroll
    for (int m = 1; m < 16; m <<= 1)
      partial[r] += __shfl_xor(partial[r], m, 32);
  }
  if (lm == 0) {
    #pragma unroll
    for (int r = 0; r < 8; ++r) s_rowpart[r + lh * 8][wave] = partial[r];
  }
  __syncthreads();
  if (tid < 16) {
    float s = 0.f;
    #pragma unroll
    for (int w = 0; w < 8; ++w) s += s_rowpart[tid][w];
    s_inv[tid] = 1.f / s;
  }
  __syncthreads();

  // normalized stores: per instruction, two 64B-aligned coalesced segments
  #pragma unroll
  for (int r = 0; r < 8; ++r) {
    const int m = r + lh * 8;
    const float inv = s_inv[m];
    float* orow = out + (size_t)(m0 + m) * kK + n_base + lm;
    #pragma unroll
    for (int t = 0; t < 8; ++t)
      orow[t * 16] = acc[t][r] * inv;
  }
}

extern "C" void kernel_launch(void* const* d_in, const int* in_sizes, int n_in,
                              void* d_out, int out_size, void* d_ws, size_t ws_size,
                              hipStream_t stream) {
  const float* z       = (const float*)d_in[0];  // [64,512,256] f32
  const float* centers = (const float*)d_in[1];  // [1024,256] f32
  float* out = (float*)d_out;                    // [64,512,1024] f32

  // workspace layout (~34.6 MB)
  char* ws = (char*)d_ws;
  _Float16* zh = (_Float16*)ws; ws += (size_t)kN * kD * sizeof(_Float16);
  _Float16* zl = (_Float16*)ws; ws += (size_t)kN * kD * sizeof(_Float16);
  _Float16* ch = (_Float16*)ws; ws += (size_t)kK * kD * sizeof(_Float16);
  _Float16* cl = (_Float16*)ws; ws += (size_t)kK * kD * sizeof(_Float16);
  float*   csq = (float*)ws;

  const int nz = kN * kD;   // 8388608
  const int nc = kK * kD;   // 262144
  cvt_split_kernel<<<(nz + 255) / 256, 256, 0, stream>>>(z, zh, zl, nz);
  cvt_split_kernel<<<(nc + 255) / 256, 256, 0, stream>>>(centers, ch, cl, nc);
  csq_kernel<<<(kK + 255) / 256, 256, 0, stream>>>(centers, csq);

  som_soft_assign_kernel<<<kN / 16, 256, 0, stream>>>(z, zh, zl, ch, cl, csq, out);
}
